// Attention_6897717478041
// MI455X (gfx1250) — matrix operand order repaired and checked by reading, not compiled, not run on hardware
//
#include <hip/hip_runtime.h>
#include <hip/hip_bf16.h>
#include <cstdint>

typedef __attribute__((ext_vector_type(16))) _Float16 v16h;
typedef __attribute__((ext_vector_type(8)))  _Float16 v8h;
typedef __attribute__((ext_vector_type(8)))  float    v8f;

union HF16 { v16h v; v8h h[2]; };

__device__ __forceinline__ v8f wmma_f16(const HF16& a, const HF16& b, v8f c) {
    return __builtin_amdgcn_wmma_f32_16x16x32_f16(false, a.v, false, b.v,
                                                  (short)0, c, false, false);
}

// CDNA5 async global->LDS copy (16B per lane), tracked by ASYNCcnt.
// LDS address comes from the vdst VGPR; generic shared pointers carry the LDS byte
// offset in their low 32 bits, so we pass that directly.
__device__ __forceinline__ void async_copy_b128(const _Float16* lds_dst,
                                                const _Float16* gsrc) {
    unsigned lds_off = (unsigned)(uintptr_t)lds_dst;
    uint64_t gaddr   = (uint64_t)(uintptr_t)gsrc;
    asm volatile("global_load_async_to_lds_b128 %0, %1, off"
                 :: "v"(lds_off), "v"(gaddr)
                 : "memory");
}

__device__ __forceinline__ void wait_async0() {
    asm volatile("s_wait_asynccnt 0x0" ::: "memory");
}

// ---------------------------------------------------------------- convert ---
__global__ void f32_to_f16_kernel(const float* __restrict__ in,
                                  _Float16* __restrict__ out, int n) {
    int i = blockIdx.x * blockDim.x + threadIdx.x;
    if (i < n) out[i] = (_Float16)in[i];
}

// -------------------------------------------------------------------- GEMM --
// C[M,N] = A[M,K] * B[K,N] + bias[N].  Tiles: 128x128x32, 256 threads (8 waves),
// wave grid 4(M) x 2(N), each wave computes 32x64 = 2x4 WMMA 16x16 tiles.
// A tile staged with global_load_async_to_lds_b128; B tile transposed manually.
template <bool OUT_F16>
__global__ __launch_bounds__(256)
void gemm_f16_kernel(const _Float16* __restrict__ A,
                     const _Float16* __restrict__ Bm,
                     const float* __restrict__ bias,
                     void* __restrict__ Cout,
                     int M, int N, int K) {
    __shared__ alignas(16) _Float16 As[128 * 40];   // row-major [row][k], stride 40
    __shared__ alignas(16) _Float16 Bs[128 * 40];   // col-major [col][k], stride 40

    const int tid  = threadIdx.x;
    const int lane = tid & 31;
    const int wid  = tid >> 5;
    const int wm   = wid >> 1;            // 0..3
    const int wn   = wid & 1;             // 0..1
    const int bm   = blockIdx.x * 128;
    const int bn   = blockIdx.y * 128;

    const int nlo = lane & 15;
    const int hi  = lane >> 4;            // 0 or 1
    const int ak  = hi ? 8 : 0;           // A-frag K base
    const int bk  = hi ? 8 : 0;           // B-frag K base

    v8f acc[2][4] = {};

    for (int k0 = 0; k0 < K; k0 += 32) {
        __syncthreads();
        // A tile: 128x32 f16, async copy 2x16B per thread into padded LDS rows
        {
            int row = tid >> 1;
            int ks  = (tid & 1) * 16;
            const _Float16* src = A + (size_t)(bm + row) * K + k0 + ks;
            _Float16*       dst = &As[row * 40 + ks];
            async_copy_b128(dst,     src);
            async_copy_b128(dst + 8, src + 8);
        }
        // B tile: 32x128 row-major in global -> column-major in LDS (transpose)
        {
            int k  = tid >> 3;
            int cs = (tid & 7) * 16;
            const _Float16* src = Bm + (size_t)(k0 + k) * N + bn + cs;
#pragma unroll
            for (int i = 0; i < 16; ++i) Bs[(cs + i) * 40 + k] = src[i];
        }
        wait_async0();
        __syncthreads();

        HF16 af[2], bf[4];
#pragma unroll
        for (int mt = 0; mt < 2; ++mt) {
            const _Float16* p = &As[(wm * 32 + mt * 16 + nlo) * 40 + ak];
            af[mt].h[0] = *(const v8h*)(p);
            af[mt].h[1] = *(const v8h*)(p + 16);
        }
#pragma unroll
        for (int nt = 0; nt < 4; ++nt) {
            const _Float16* p = &Bs[(wn * 64 + nt * 16 + nlo) * 40 + bk];
            bf[nt].h[0] = *(const v8h*)(p);
            bf[nt].h[1] = *(const v8h*)(p + 16);
        }
#pragma unroll
        for (int mt = 0; mt < 2; ++mt)
#pragma unroll
            for (int nt = 0; nt < 4; ++nt)
                acc[mt][nt] = wmma_f16(af[mt], bf[nt], acc[mt][nt]);
    }

    // epilogue: C/D layout -> global
    const int row0 = bm + wm * 32;
    const int col0 = bn + wn * 64;
#pragma unroll
    for (int mt = 0; mt < 2; ++mt) {
#pragma unroll
        for (int nt = 0; nt < 4; ++nt) {
            int col = col0 + nt * 16 + nlo;
            float bv = bias[col];
#pragma unroll
            for (int r = 0; r < 8; ++r) {
                int row = row0 + mt * 16 + hi * 8 + r;
                float val = acc[mt][nt][r] + bv;
                if (OUT_F16)
                    ((_Float16*)Cout)[(size_t)row * N + col] = (_Float16)val;
                else
                    ((float*)Cout)[(size_t)row * N + col] = val;
            }
        }
    }
}

// --------------------------------------------------------- flash attention --
// qkv: f16 [B][N][3*D]; out: f16 [B][N][D] (head-interleaved).
// One workgroup per (b, h, 128-query block); 8 waves, wave owns 16 query rows.
__global__ __launch_bounds__(256)
void attn_kernel(const _Float16* __restrict__ qkv, _Float16* __restrict__ out) {
    constexpr int N  = 1024;
    constexpr int D3 = 2304;
    constexpr int H  = 12;
    constexpr int HD = 64;
    constexpr int DO = 768;

    __shared__ alignas(16) _Float16 Vt[64 * 40];      // [hd][key], stride 40
    __shared__ alignas(16) _Float16 Pl[8][16 * 40];   // per-wave P tile

    const int tid  = threadIdx.x;
    const int lane = tid & 31;
    const int wid  = tid >> 5;
    const int nlo  = lane & 15;
    const int hi   = lane >> 4;
    const int ak   = hi ? 8 : 0;
    const int bk   = hi ? 8 : 0;

    const int qblk = blockIdx.x & 7;
    const int bh   = blockIdx.x >> 3;
    const int b    = bh / H;
    const int h    = bh % H;

    const _Float16* Qb = qkv + (size_t)b * N * D3 + h * HD;
    const _Float16* Kb = Qb + 768;
    const _Float16* Vb = Qb + 1536;

    // Q A-fragments (scale 1/sqrt(64) = 0.125 folded in)
    const int qrow = qblk * 128 + wid * 16 + nlo;
    const _Float16* qp = Qb + (size_t)qrow * D3;
    HF16 aq[2];
#pragma unroll
    for (int kt = 0; kt < 2; ++kt) {
        aq[kt].h[0] = *(const v8h*)(qp + kt * 32 + ak);
        aq[kt].h[1] = *(const v8h*)(qp + kt * 32 + ak + 16);
#pragma unroll
        for (int i = 0; i < 16; ++i)
            aq[kt].v[i] = aq[kt].v[i] * (_Float16)0.125f;
    }

    v8f o[4] = {};
    float m[8], l[8];
#pragma unroll
    for (int r = 0; r < 8; ++r) { m[r] = -3.0e38f; l[r] = 0.0f; }

    for (int kb0 = 0; kb0 < N; kb0 += 32) {
        __syncthreads();
        // stage V block (32 keys x 64 hd) transposed into LDS
        {
            int key = tid >> 3;
            int hd0 = (tid & 7) * 8;
            const _Float16* vp = Vb + (size_t)(kb0 + key) * D3 + hd0;
#pragma unroll
            for (int i = 0; i < 8; ++i) Vt[(hd0 + i) * 40 + key] = vp[i];
        }
        // warm L2/WGP$ for next key block (K rows reused by 8 waves x 8 blocks)
        {
            int nkb = (kb0 + 32 < N) ? kb0 + 32 : kb0;
            __builtin_prefetch(Kb + (size_t)(nkb + (tid & 31)) * D3, 0, 0);
            __builtin_prefetch(Vb + (size_t)(nkb + (tid & 31)) * D3, 0, 0);
        }
        __syncthreads();

        // S = Q*K^T for 2 subtiles of 16 keys (B-frags loaded straight from global)
        v8f s0 = {}, s1 = {};
        {
            const _Float16* kp = Kb + (size_t)(kb0 + nlo) * D3;
            HF16 b0, b1;
            b0.h[0] = *(const v8h*)(kp + bk);
            b0.h[1] = *(const v8h*)(kp + bk + 16);
            b1.h[0] = *(const v8h*)(kp + 32 + bk);
            b1.h[1] = *(const v8h*)(kp + 32 + bk + 16);
            s0 = wmma_f16(aq[0], b0, s0);
            s0 = wmma_f16(aq[1], b1, s0);
        }
        {
            const _Float16* kp = Kb + (size_t)(kb0 + 16 + nlo) * D3;
            HF16 b0, b1;
            b0.h[0] = *(const v8h*)(kp + bk);
            b0.h[1] = *(const v8h*)(kp + bk + 16);
            b1.h[0] = *(const v8h*)(kp + 32 + bk);
            b1.h[1] = *(const v8h*)(kp + 32 + bk + 16);
            s1 = wmma_f16(aq[0], b0, s1);
            s1 = wmma_f16(aq[1], b1, s1);
        }

        // online softmax; rows live in 16-lane halves -> shfl_xor(width 16)
        float corr[8];
#pragma unroll
        for (int r = 0; r < 8; ++r) {
            float mx = fmaxf(s0[r], s1[r]);
#pragma unroll
            for (int off = 1; off < 16; off <<= 1)
                mx = fmaxf(mx, __shfl_xor(mx, off, 16));
            float mnew = fmaxf(m[r], mx);
            float c    = __expf(m[r] - mnew);
            float p0   = __expf(s0[r] - mnew);
            float p1   = __expf(s1[r] - mnew);
            float rs   = p0 + p1;
#pragma unroll
            for (int off = 1; off < 16; off <<= 1)
                rs += __shfl_xor(rs, off, 16);
            l[r]  = l[r] * c + rs;
            m[r]  = mnew;
            corr[r] = c;
            _Float16* pp = &Pl[wid][(r + hi * 8) * 40];
            pp[nlo]      = (_Float16)p0;   // keys kb0 + nlo
            pp[16 + nlo] = (_Float16)p1;   // keys kb0 + 16 + nlo
        }

        // rescale O, then O += P * V
#pragma unroll
        for (int d = 0; d < 4; ++d)
#pragma unroll
            for (int r = 0; r < 8; ++r) o[d][r] *= corr[r];

        HF16 pf;
        {
            const _Float16* p = &Pl[wid][nlo * 40 + ak];
            pf.h[0] = *(const v8h*)(p);
            pf.h[1] = *(const v8h*)(p + 16);
        }
#pragma unroll
        for (int d = 0; d < 4; ++d) {
            const _Float16* vp = &Vt[(d * 16 + nlo) * 40 + bk];
            HF16 bv;
            bv.h[0] = *(const v8h*)(vp);
            bv.h[1] = *(const v8h*)(vp + 16);
            o[d] = wmma_f16(pf, bv, o[d]);
        }
    }

    // finalize: divide by l, write f16 head-interleaved output
    float inv[8];
#pragma unroll
    for (int r = 0; r < 8; ++r) inv[r] = 1.0f / l[r];
    _Float16* ob = out + ((size_t)b * N + qblk * 128 + wid * 16) * DO + h * HD;
#pragma unroll
    for (int d = 0; d < 4; ++d)
#pragma unroll
        for (int r = 0; r < 8; ++r)
            ob[(size_t)(r + hi * 8) * DO + d * 16 + nlo] =
                (_Float16)(o[d][r] * inv[r]);
}

// ------------------------------------------------------------------ launch --
extern "C" void kernel_launch(void* const* d_in, const int* in_sizes, int n_in,
                              void* d_out, int out_size, void* d_ws, size_t ws_size,
                              hipStream_t stream) {
    const float* x      = (const float*)d_in[0];
    const float* w_qkv  = (const float*)d_in[1];
    const float* b_qkv  = (const float*)d_in[2];
    const float* w_proj = (const float*)d_in[3];
    const float* b_proj = (const float*)d_in[4];

    constexpr int B = 16, N = 1024, D = 768;
    constexpr int M  = B * N;          // 16384
    constexpr int D3 = 3 * D;          // 2304

    char* ws = (char*)d_ws;
    _Float16* xh     = (_Float16*)(ws);                         // 16384*768
    _Float16* wqkvh  = (_Float16*)(ws + 25165824);              // 768*2304
    _Float16* wprojh = (_Float16*)(ws + 28704768);              // 768*768
    _Float16* qkvh   = (_Float16*)(ws + 29884416);              // 16384*2304
    _Float16* attnh  = (_Float16*)(ws + 105381888);             // 16384*768

    // converts
    {
        int n = M * D;
        f32_to_f16_kernel<<<(n + 255) / 256, 256, 0, stream>>>(x, xh, n);
    }
    {
        int n = D * D3;
        f32_to_f16_kernel<<<(n + 255) / 256, 256, 0, stream>>>(w_qkv, wqkvh, n);
    }
    {
        int n = D * D;
        f32_to_f16_kernel<<<(n + 255) / 256, 256, 0, stream>>>(w_proj, wprojh, n);
    }

    // qkv = x @ w_qkv + b_qkv  -> f16
    {
        dim3 grid(M / 128, D3 / 128);
        gemm_f16_kernel<true><<<grid, 256, 0, stream>>>(xh, wqkvh, b_qkv,
                                                        (void*)qkvh, M, D3, D);
    }

    // flash attention
    {
        dim3 grid(B * 12 * (N / 128));
        attn_kernel<<<grid, 256, 0, stream>>>(qkvh, attnh);
    }

    // out = attn @ w_proj + b_proj -> f32
    {
        dim3 grid(M / 128, D / 128);
        gemm_f16_kernel<false><<<grid, 256, 0, stream>>>(attnh, wprojh, b_proj,
                                                         d_out, M, D, D);
    }
}
